// ResidualBlock_89764816486700
// MI455X (gfx1250) — matrix operand order repaired and checked by reading, not compiled, not run on hardware
//
#include <hip/hip_runtime.h>
#include <hip/hip_bf16.h>

typedef __bf16 bf16;
typedef __bf16 v16bf  __attribute__((ext_vector_type(16)));
typedef __bf16 bf16x8 __attribute__((ext_vector_type(8)));
typedef float  v8f    __attribute__((ext_vector_type(8)));
typedef int    v4i    __attribute__((ext_vector_type(4)));

#define B_   64
#define C_   128
#define H_   56
#define W_   56
#define HW_  (H_ * W_)
#define KDIM 1152                       // C_ * 9, k = (kh*3+kw)*128 + c
#define NPIX ((size_t)B_ * HW_)         // NHWC rows
#define NELEM ((size_t)B_ * C_ * HW_)   // total elements

// ---- gfx1250 async global->LDS path (guarded; fallback = sync copy) --------
#if defined(__has_builtin)
#  if __has_builtin(__builtin_amdgcn_global_load_async_to_lds_b128)
#    define HAVE_ASYNC_LDS 1
#  endif
#endif
#ifndef HAVE_ASYNC_LDS
#  define HAVE_ASYNC_LDS 0
#endif

__device__ __forceinline__ void async_cp16(const bf16* g, bf16* l) {
#if HAVE_ASYNC_LDS
  __builtin_amdgcn_global_load_async_to_lds_b128(
      (__attribute__((address_space(1))) v4i*)(g),
      (__attribute__((address_space(3))) v4i*)(l), 0, 0);
#else
  *(bf16x8*)l = *(const bf16x8*)g;
#endif
}

template <int N>
__device__ __forceinline__ void wait_async() {
#if defined(__has_builtin) && __has_builtin(__builtin_amdgcn_s_wait_asynccnt)
  __builtin_amdgcn_s_wait_asynccnt(N);
#elif HAVE_ASYNC_LDS
  asm volatile("s_wait_asynccnt %0" ::"i"(N));
#endif
}

// ---------------------------------------------------------------- prep kernels

// NCHW f32 -> NHWC bf16
__global__ void k_prep_x(const float* __restrict__ x, bf16* __restrict__ xb) {
  size_t stride = (size_t)gridDim.x * blockDim.x;
  for (size_t i = (size_t)blockIdx.x * blockDim.x + threadIdx.x; i < NELEM; i += stride) {
    size_t p = i % HW_;
    size_t c = (i / HW_) % C_;
    size_t b = i / ((size_t)HW_ * C_);
    xb[(b * HW_ + p) * C_ + c] = (bf16)x[i];
  }
}

// [b][o][c][kh][kw] f32 -> [b][o][r*128+c] bf16  (r = kh*3+kw)
__global__ void k_prep_w(const float* __restrict__ w, bf16* __restrict__ wb) {
  size_t total = (size_t)B_ * C_ * KDIM;
  size_t stride = (size_t)gridDim.x * blockDim.x;
  for (size_t j = (size_t)blockIdx.x * blockDim.x + threadIdx.x; j < total; j += stride) {
    size_t c = j & 127;
    size_t r = (j >> 7) % 9;
    size_t o = (j / KDIM) % C_;
    size_t b = j / ((size_t)KDIM * C_);
    wb[j] = (bf16)w[(((b * C_ + o) * C_ + c) * 9) + r];
  }
}

__global__ void k_zero(float* __restrict__ s, int n) {
  for (int i = threadIdx.x; i < n; i += blockDim.x) s[i] = 0.f;
}

// ---------------------------------------------------------------- implicit-GEMM conv
// act: NHWC bf16, wgt: [b][o][k=1152] bf16, y: NHWC f32
// grid = (H_, B_), block = 256 (8 waves). Block tile: M=128 x N=64 (one image row).
// 36 K-steps: kh outer (3), fully-unrolled q = cc*3+kw inner (12).
// A (weights): 3-deep ring buffer, distance-2 async prefetch, wait asynccnt<=2.
// B: zero-padded activation row [66 px][128 ch], double-buffered by kh parity.
// Exactly ONE barrier per K-step.
__global__ __launch_bounds__(256)
void k_conv(const bf16* __restrict__ act, const bf16* __restrict__ wgt,
            float* __restrict__ y) {
  __shared__ alignas(32) bf16 ldsA[3][128 * 32];   // [m][k]   8 KB x3
  __shared__ alignas(32) bf16 ldsB[2][66 * 128];   // [px][ch] 16.5 KB x2, px = w+1

  const int h  = blockIdx.x;
  const int b  = blockIdx.y;
  const int t  = threadIdx.x;
  const int lane  = t & 31;
  const int wv    = t >> 5;           // wave id 0..7
  const int mBase = wv * 16;          // this wave's 16 output channels
  const int nn = lane & 15;
  const int hi = lane >> 4;

  v8f acc[4];
  #pragma unroll
  for (int j = 0; j < 4; ++j)
    #pragma unroll
    for (int v = 0; v < 8; ++v) acc[j][v] = 0.f;

  // cooperative-load roles
  const int aO    = t >> 1;           // 0..127 : A row (out channel)
  const int aHalf = t & 1;            // 0/1    : which 16-element half of K

  const bf16* wB   = wgt + (size_t)b * C_ * KDIM;
  const bf16* actB = act + (size_t)b * HW_ * C_;

  const bf16 z = (bf16)0.f;
  const bf16x8 bzero8 = {z, z, z, z, z, z, z, z};

  // A stage: 2 unconditional async 16B copies per thread (exact per-wave count = 2)
  auto stageA = [&](int r, int c0, int buf) {
    const bf16* gA = wB + (size_t)aO * KDIM + r * 128 + c0 + aHalf * 16;
    bf16* lA = &ldsA[buf][aO * 32 + aHalf * 16];
    async_cp16(gA, lA);
    async_cp16(gA + 8, lA + 8);
  };

  // B stage: padded row hh = h+kh-1 -> ldsB[bufB][px][ch], px = w+1 (0,57 = zero halo)
  // 58 px * 16 chunks of 8 ch = 928 16B chunks, <=4 per thread; zeros via ds_store
  auto stageB = [&](int kh, int bufB) {
    const int hh = h + kh - 1;
    const bool rowok = (unsigned)hh < H_;
    #pragma unroll
    for (int i = 0; i < 4; ++i) {
      const int idx = t + i * 256;
      if (idx < 928) {
        const int px  = idx >> 4;
        const int ch8 = (idx & 15) * 8;
        bf16* dst = &ldsB[bufB][px * 128 + ch8];
        if (rowok && px >= 1 && px <= 56)
          async_cp16(actB + ((size_t)hh * W_ + (px - 1)) * C_ + ch8, dst);
        else
          *(bf16x8*)dst = bzero8;
      }
    }
  };

  // prologue: B(kh=0) first, then A steps 0 and 1 (newest ops must be a full A stage)
  stageB(0, 0);
  stageA(0, 0, 0);          // s=0: kh0 q0 -> r=0,  c0=0,  buf 0
  stageA(1, 0, 1);          // s=1: kh0 q1 -> r=1,  c0=0,  buf 1

  for (int kh = 0; kh < 3; ++kh) {
    const int bB = kh & 1;
    #pragma unroll
    for (int q = 0; q < 12; ++q) {          // q = cc*3 + kw (all constants when unrolled)
      const int kw   = q % 3;
      const int c0   = (q / 3) * 32;
      const int bufA = q % 3;               // == (kh*12+q) % 3
      if (kh == 2 && q == 11) wait_async<0>();
      else                    wait_async<2>();   // all but newest A stage landed
      __syncthreads();                      // every wave's stage-s data visible

      // A fragment (16x32): lane(m=nn,hi): K = {16hi..16hi+15}
      union { v16bf v; bf16x8 hv[2]; } af;
      const bf16x8* aBase = (const bf16x8*)&ldsA[bufA][0];
      af.hv[0] = aBase[(mBase + nn) * 4 + 2 * hi];
      af.hv[1] = aBase[(mBase + nn) * 4 + 2 * hi + 1];

      // B fragments: contiguous 32B reads at immediate offsets, halo zeros in LDS
      v16bf bfr[4];
      #pragma unroll
      for (int j = 0; j < 4; ++j)
        bfr[j] = *(const v16bf*)&ldsB[bB][(j * 16 + nn + kw) * 128 + c0 + hi * 16];

      #pragma unroll
      for (int j = 0; j < 4; ++j)
        acc[j] = __builtin_amdgcn_wmma_f32_16x16x32_bf16(
            false, af.v, false, bfr[j], (short)0, acc[j], false, false);

      // post-compute issues (safe: we passed this step's barrier).
      // B first so the exact wait<2> (newest 2 = next A stage) stays valid.
      if (q == 11 && kh < 2) stageB(kh + 1, (kh + 1) & 1);
      if (!(kh == 2 && q >= 10)) {          // issue A stage s+2
        int nq = q + 2, nkh = kh;
        if (nq >= 12) { nq -= 12; ++nkh; }
        stageA(nkh * 3 + (nq % 3), (nq / 3) * 32, nq % 3);
      }
    }
  }

  // store: C/D layout M = v + 8*hi -> 8 contiguous channels per lane (32B store)
  float* yrow = y + ((size_t)b * H_ + h) * (size_t)W_ * C_;
  #pragma unroll
  for (int j = 0; j < 4; ++j) {
    const int p = j * 16 + nn;
    if (p < W_) {
      *(v8f*)(yrow + (size_t)p * C_ + mBase + hi * 8) = acc[j];
    }
  }
}

// ---------------------------------------------------------------- batchnorm
__global__ __launch_bounds__(256)
void k_bnstats(const float* __restrict__ y, float* __restrict__ sum,
               float* __restrict__ sumsq) {
  __shared__ float s1[256], s2[256];
  const int t = threadIdx.x;
  const int c = t & 127;
  const int half = t >> 7;
  const size_t rowsPerBlk = NPIX / gridDim.x;        // 200704/512 = 392
  const size_t row0 = (size_t)blockIdx.x * rowsPerBlk;
  float a = 0.f, q = 0.f;
  for (size_t i = half; i < rowsPerBlk; i += 2) {
    float v = y[(row0 + i) * C_ + c];                 // coalesced: c contiguous
    a += v; q += v * v;
  }
  s1[t] = a; s2[t] = q;
  __syncthreads();
  if (t < 128) {
    atomicAdd(&sum[c],   s1[t] + s1[t + 128]);
    atomicAdd(&sumsq[c], s2[t] + s2[t + 128]);
  }
}

__global__ void k_bnfinal(const float* __restrict__ sum, const float* __restrict__ sumsq,
                          float* __restrict__ mean, float* __restrict__ rstd) {
  int c = threadIdx.x;
  if (c < C_) {
    const float n = (float)NPIX;
    float m = sum[c] / n;
    float v = sumsq[c] / n - m * m;
    mean[c] = m;
    rstd[c] = rsqrtf(v + 1e-5f);
  }
}

// y1 NHWC f32 -> BN+ReLU -> bf16 NHWC
__global__ void k_bnrelu_bf16(const float* __restrict__ y, const float* __restrict__ mean,
                              const float* __restrict__ rstd, const float* __restrict__ g,
                              const float* __restrict__ be, bf16* __restrict__ out) {
  size_t stride = (size_t)gridDim.x * blockDim.x;
  for (size_t i = (size_t)blockIdx.x * blockDim.x + threadIdx.x; i < NELEM; i += stride) {
    int c = (int)(i & 127);
    float v = (y[i] - mean[c]) * rstd[c] * g[c] + be[c];
    out[i] = (bf16)fmaxf(v, 0.f);
  }
}

// y2 NHWC f32 -> BN + residual(x NCHW) + ReLU -> out NCHW f32
__global__ void k_out(const float* __restrict__ y, const float* __restrict__ mean,
                      const float* __restrict__ rstd, const float* __restrict__ g,
                      const float* __restrict__ be, const float* __restrict__ x,
                      float* __restrict__ out) {
  size_t stride = (size_t)gridDim.x * blockDim.x;
  for (size_t i = (size_t)blockIdx.x * blockDim.x + threadIdx.x; i < NELEM; i += stride) {
    size_t p = i % HW_;
    size_t c = (i / HW_) % C_;
    size_t b = i / ((size_t)HW_ * C_);
    float v = (y[(b * HW_ + p) * C_ + c] - mean[c]) * rstd[c] * g[c] + be[c];
    out[i] = fmaxf(v + x[i], 0.f);
  }
}

// ---------------------------------------------------------------- launch
extern "C" void kernel_launch(void* const* d_in, const int* in_sizes, int n_in,
                              void* d_out, int out_size, void* d_ws, size_t ws_size,
                              hipStream_t stream) {
  const float* x   = (const float*)d_in[0];
  const float* w1  = (const float*)d_in[1];
  const float* w2  = (const float*)d_in[2];
  const float* g1  = (const float*)d_in[3];
  const float* be1 = (const float*)d_in[4];
  const float* g2  = (const float*)d_in[5];
  const float* be2 = (const float*)d_in[6];
  float* out = (float*)d_out;

  char* ws = (char*)d_ws;
  const size_t MiB = 1024 * 1024;
  bf16*  xb    = (bf16*)(ws + 0);            //  49 MiB: NHWC bf16 input
  bf16*  w1b   = (bf16*)(ws + 49  * MiB);    //  18 MiB
  bf16*  w2b   = (bf16*)(ws + 67  * MiB);    //  18 MiB
  float* y1    = (float*)(ws + 85  * MiB);   //  98 MiB: conv1 out NHWC f32
  bf16*  a1b   = (bf16*)(ws + 183 * MiB);    //  49 MiB: act1 NHWC bf16
  float* y2    = (float*)(ws + 232 * MiB);   //  98 MiB: conv2 out NHWC f32
  float* stats = (float*)(ws + 330 * MiB);   //  8 * 128 floats
  float* sum1 = stats,        *sq1 = stats + 128,  *mean1 = stats + 256, *rstd1 = stats + 384;
  float* sum2 = stats + 512,  *sq2 = stats + 640,  *mean2 = stats + 768, *rstd2 = stats + 896;

  dim3 cgrid(H_, B_);

  k_zero<<<1, 256, 0, stream>>>(stats, 1024);
  k_prep_x<<<4096, 256, 0, stream>>>(x, xb);
  k_prep_w<<<4096, 256, 0, stream>>>(w1, w1b);
  k_prep_w<<<4096, 256, 0, stream>>>(w2, w2b);

  k_conv<<<cgrid, 256, 0, stream>>>(xb, w1b, y1);
  k_bnstats<<<512, 256, 0, stream>>>(y1, sum1, sq1);
  k_bnfinal<<<1, 128, 0, stream>>>(sum1, sq1, mean1, rstd1);
  k_bnrelu_bf16<<<4096, 256, 0, stream>>>(y1, mean1, rstd1, g1, be1, a1b);

  k_conv<<<cgrid, 256, 0, stream>>>(a1b, w2b, y2);
  k_bnstats<<<512, 256, 0, stream>>>(y2, sum2, sq2);
  k_bnfinal<<<1, 128, 0, stream>>>(sum2, sq2, mean2, rstd2);
  k_out<<<4096, 256, 0, stream>>>(y2, mean2, rstd2, g2, be2, x, out);
}
